// TIGraNet_mnist_rot_15513421873217
// MI455X (gfx1250) — matrix operand order repaired and checked by reading, not compiled, not run on hardware
//
#include <hip/hip_runtime.h>

// ---------------- problem constants ----------------
#define NN     784      // 28*28 nodes
#define NP     800      // padded node count: 50*16 (M) and 25*32 (K)
#define BATCH  512
#define F1     10       // conv1 output filters
#define F2     20       // conv2 output filters
#define NTERMS 5        // spectral conv polynomial terms (k=0..4)
#define NCHEB  14       // Chebyshev statistic terms (k=0..13)
#define TOPK1  600
#define TOPK2  300

typedef __bf16 bf16_t;
typedef bf16_t v16bf __attribute__((ext_vector_type(16)));
typedef bf16_t v8bf  __attribute__((ext_vector_type(8)));
typedef float  v8f   __attribute__((ext_vector_type(8)));

// =====================================================================
// WMMA GEMM:  Y[c, m] = sum_k A[m, k] * X[c, k]    (all node dims = NP)
//   A : NP x NP row-major bf16 (Laplacian, zero padded outside 784x784)
//   X : C  x NP bf16  (activation, node index contiguous per column c)
//   Y : C  x NP bf16
//   Tprev != nullptr  =>  Y = 2*acc - Tprev   (Chebyshev step; Y may
//                         alias Tprev: each element is read exactly once
//                         by the lane that writes it)
// 256-thread block = 8 waves; each wave computes a 32(m) x 64(c) tile
// with 8 WMMA accumulators -> 8 v_wmma per 12 b128 loads.
// =====================================================================

// A fragment, 16x32 bf16: lane holds row (lane&15), contiguous K run
// [half*16 .. half*16+15]  -> two 16B loads.
__device__ __forceinline__ v16bf load_a_frag(const bf16_t* p) {
  v8bf lo = *(const v8bf*)(p);
  v8bf hi = *(const v8bf*)(p + 8);
  v16bf r;
#pragma unroll
  for (int i = 0; i < 8; ++i) { r[i] = lo[i]; r[i + 8] = hi[i]; }
  return r;
}

__device__ __forceinline__ void store_d_tile(bf16_t* Y, const bf16_t* Tprev,
                                             v8f acc, int c, int mbase) {
  size_t off = (size_t)c * NP + mbase;
  v8bf out;
  if (Tprev) {
    v8bf tp = *(const v8bf*)(Tprev + off);
#pragma unroll
    for (int r = 0; r < 8; ++r) out[r] = (bf16_t)(2.0f * acc[r] - (float)tp[r]);
  } else {
#pragma unroll
    for (int r = 0; r < 8; ++r) out[r] = (bf16_t)acc[r];
  }
  *(v8bf*)(Y + off) = out;
}

__global__ __launch_bounds__(256) void k_gemm_wmma(
    const bf16_t* __restrict__ A, const bf16_t* __restrict__ X,
    bf16_t* Y, const bf16_t* Tprev, int C) {
  const int lane = threadIdx.x & 31;
  const int wave = threadIdx.x >> 5;
  const int row  = lane & 15;
  const int half = lane >> 4;
  const int c0 = blockIdx.x * 512 + wave * 64;  // this wave's 64-column slice
  const int m0 = blockIdx.y * 32;               // 32-node slice

  const bf16_t* a0 = A + (size_t)(m0 + row) * NP + half * 16;
  const bf16_t* a1 = a0 + (size_t)16 * NP;
  // B fragment 32x16 bf16: lane holds column (lane&15), contiguous
  // K run [half*16 .. half*16+15] -> one 32B load.
  const bf16_t* b[4];
  b[0] = X + (size_t)(c0 + row) * NP + half * 16;
#pragma unroll
  for (int j = 1; j < 4; ++j) b[j] = b[j - 1] + (size_t)16 * NP;

  const v8f vzero = {};
  v8f acc0[4], acc1[4];
#pragma unroll
  for (int j = 0; j < 4; ++j) { acc0[j] = vzero; acc1[j] = vzero; }

#pragma unroll 1
  for (int k = 0; k < NP; k += 32) {
    v16bf fa0 = load_a_frag(a0 + k);
    v16bf fa1 = load_a_frag(a1 + k);
#pragma unroll
    for (int j = 0; j < 4; ++j) {
      v16bf fb = *(const v16bf*)(b[j] + k);
      // prefetch the streamed operand 256B ahead (-> global_prefetch_b8)
      __builtin_prefetch((const void*)(b[j] + k + 128), 0, 1);
      acc0[j] = __builtin_amdgcn_wmma_f32_16x16x32_bf16(false, fa0, false, fb,
                                                        (short)0, acc0[j], false, false);
      acc1[j] = __builtin_amdgcn_wmma_f32_16x16x32_bf16(false, fa1, false, fb,
                                                        (short)0, acc1[j], false, false);
    }
  }

  // D layout: lane holds column N=(lane&15); VGPR r holds m = 8*half + r.
#pragma unroll
  for (int j = 0; j < 4; ++j) {
    store_d_tile(Y, Tprev, acc0[j], c0 + j * 16 + row, m0 + half * 8);
    store_d_tile(Y, Tprev, acc1[j], c0 + j * 16 + row, m0 + 16 + half * 8);
  }
}

// ---------------- bf16 convert + pad of the 784x784 Laplacians ----------------
__global__ void k_convert_pad(const float* __restrict__ src, bf16_t* __restrict__ dst) {
  int idx = blockIdx.x * blockDim.x + threadIdx.x;
  if (idx >= NP * NP) return;
  int m = idx / NP, k = idx % NP;
  float v = (m < NN && k < NN) ? src[m * NN + k] : 0.0f;
  dst[idx] = (bf16_t)v;
}

// ---------------- zero-mean input, to bf16 (C=BATCH x NP layout) ----------------
__global__ __launch_bounds__(256) void k_prep(const float* __restrict__ in,
                                              bf16_t* __restrict__ x0) {
  __shared__ float red[256];
  int b = blockIdx.x, t = threadIdx.x;
  float s = 0.0f;
  for (int n = t; n < NN; n += 256) s += in[b * NN + n];
  red[t] = s; __syncthreads();
  for (int w = 128; w > 0; w >>= 1) { if (t < w) red[t] += red[t + w]; __syncthreads(); }
  float mean = red[0] * (1.0f / NN);
  for (int n = t; n < NP; n += 256)
    x0[(size_t)b * NP + n] = (bf16_t)((n < NN) ? (in[b * NN + n] - mean) : 0.0f);
}

// ---------------- accumulate alpha projection: acc[b,h,n] += sum_g yk[b,g,n]*alpha[g,h]
__global__ void k_project(const bf16_t* __restrict__ src, const float* __restrict__ alphak,
                          float* __restrict__ acc, int Fin, int Fout, int first) {
  int idx = blockIdx.x * blockDim.x + threadIdx.x;
  if (idx >= BATCH * Fout * NN) return;
  int n = idx % NN; int t = idx / NN; int h = t % Fout; int b = t / Fout;
  float s = 0.0f;
  for (int g = 0; g < Fin; ++g)
    s += (float)src[(size_t)(b * Fin + g) * NP + n] * alphak[g * Fout + h];
  size_t o = (size_t)(b * Fout + h) * NP + n;
  acc[o] = first ? s : (acc[o] + s);
}

// ---------------- bias + relu + mask, write bf16 with zero padding ----------------
__global__ void k_finish(const float* __restrict__ acc, const float* __restrict__ beta,
                         const float* __restrict__ mask, bf16_t* __restrict__ out, int Fout) {
  int idx = blockIdx.x * blockDim.x + threadIdx.x;
  if (idx >= BATCH * Fout * NP) return;
  int n = idx % NP; int t = idx / NP; int h = t % Fout; int b = t / Fout;
  float v = 0.0f;
  if (n < NN) {
    v = acc[(size_t)(b * Fout + h) * NP + n] + beta[h];
    v = fmaxf(v, 0.0f);
    if (mask) v *= mask[b * NN + n];
  }
  out[(size_t)(b * Fout + h) * NP + n] = (bf16_t)v;
}

// ---------------- pooling score: max |act| over filters (times mask) ----------------
__global__ void k_score(const bf16_t* __restrict__ sc, const float* __restrict__ mask,
                        int F, float* __restrict__ score) {
  int idx = blockIdx.x * blockDim.x + threadIdx.x;
  if (idx >= BATCH * NN) return;
  int n = idx % NN, b = idx / NN;
  float s = 0.0f;
  for (int h = 0; h < F; ++h)
    s = fmaxf(s, fabsf((float)sc[(size_t)(b * F + h) * NP + n]));
  if (mask) s *= mask[b * NN + n];
  score[idx] = s;
}

// ---------------- exact top-k mask (tie-break by lower index, like lax.top_k) ------
__global__ __launch_bounds__(800) void k_topk(const float* __restrict__ score,
                                              const float* __restrict__ oldmask,
                                              int K, float* __restrict__ newmask) {
  __shared__ float s[NN];
  int b = blockIdx.x, t = threadIdx.x;
  if (t < NN) s[t] = score[b * NN + t];
  __syncthreads();
  if (t < NN) {
    float si = s[t];
    int rank = 0;
    for (int j = 0; j < NN; ++j) {
      float sj = s[j];
      rank += (sj > si) || (sj == si && j < t);
    }
    float m = (rank < K) ? 1.0f : 0.0f;
    if (oldmask) m *= oldmask[b * NN + t];
    newmask[b * NN + t] = m;
  }
}

// ---------------- T0 = sc2 * mask2 (bf16, zero padded) ----------------
__global__ void k_cheb_init(const bf16_t* __restrict__ sc2, const float* __restrict__ mask2,
                            bf16_t* __restrict__ T0) {
  int idx = blockIdx.x * blockDim.x + threadIdx.x;
  if (idx >= BATCH * F2 * NP) return;
  int n = idx % NP; int t = idx / NP; int b = t / F2;
  float v = 0.0f;
  if (n < NN) v = (float)sc2[idx] * mask2[b * NN + n];
  T0[idx] = (bf16_t)v;
}

// ---------------- Chebyshev moments: s1 = sum_n T, s2 = sum_n T^2 ----------------
__global__ __launch_bounds__(256) void k_moments(const bf16_t* __restrict__ T,
                                                 float* __restrict__ stats, int kterm) {
  __shared__ float r1[256], r2[256];
  int bf = blockIdx.x;              // b*F2 + f
  int t = threadIdx.x;
  float s1 = 0.0f, s2 = 0.0f;
  for (int n = t; n < NN; n += 256) {
    float v = (float)T[(size_t)bf * NP + n];
    s1 += v; s2 += v * v;
  }
  r1[t] = s1; r2[t] = s2; __syncthreads();
  for (int w = 128; w > 0; w >>= 1) {
    if (t < w) { r1[t] += r1[t + w]; r2[t] += r2[t + w]; }
    __syncthreads();
  }
  if (t == 0) {
    int b = bf / F2, f = bf % F2;
    stats[b * 560 + kterm * 40 + f]      = r1[0];
    stats[b * 560 + kterm * 40 + 20 + f] = r2[0];
  }
}

// ---------------- small dense layers of the MLP head ----------------
__global__ void k_linear(const float* __restrict__ in, const float* __restrict__ W,
                         const float* __restrict__ bias, float* __restrict__ out,
                         int I, int O, int relu) {
  int idx = blockIdx.x * blockDim.x + threadIdx.x;
  if (idx >= BATCH * O) return;
  int o = idx % O, b = idx / O;
  float s = bias[o];
  for (int i = 0; i < I; ++i) s += in[b * I + i] * W[i * O + o];
  if (relu) s = fmaxf(s, 0.0f);
  out[idx] = s;
}

// =====================================================================
// Host launch
// =====================================================================
static inline unsigned blocks_for(long long n, int t) {
  return (unsigned)((n + t - 1) / t);
}

extern "C" void kernel_launch(void* const* d_in, const int* in_sizes, int n_in,
                              void* d_out, int out_size, void* d_ws, size_t ws_size,
                              hipStream_t stream) {
  (void)in_sizes; (void)n_in; (void)out_size; (void)ws_size;
  const float* input  = (const float*)d_in[0];
  const float* Lmat   = (const float*)d_in[1];
  const float* Lsmat  = (const float*)d_in[2];
  const float* alpha1 = (const float*)d_in[3];
  const float* beta1  = (const float*)d_in[4];
  const float* alpha2 = (const float*)d_in[5];
  const float* beta2  = (const float*)d_in[6];
  const float* W1 = (const float*)d_in[7];  const float* b1 = (const float*)d_in[8];
  const float* W2 = (const float*)d_in[9];  const float* b2 = (const float*)d_in[10];
  const float* W3 = (const float*)d_in[11]; const float* b3 = (const float*)d_in[12];
  const float* W4 = (const float*)d_in[13]; const float* b4 = (const float*)d_in[14];
  float* out = (float*)d_out;

  // -------- workspace carve-up (all offsets 256B aligned) --------
  char* ws = (char*)d_ws;
  size_t off = 0;
  auto alloc = [&](size_t bytes) -> void* {
    void* p = ws + off;
    off += (bytes + 255) & ~(size_t)255;
    return p;
  };
  bf16_t* Lbf   = (bf16_t*)alloc((size_t)NP * NP * 2);
  bf16_t* Lsbf  = (bf16_t*)alloc((size_t)NP * NP * 2);
  bf16_t* x0    = (bf16_t*)alloc((size_t)BATCH * NP * 2);
  bf16_t* yA    = (bf16_t*)alloc((size_t)BATCH * F2 * NP * 2);   // max width C=10240
  bf16_t* yB    = (bf16_t*)alloc((size_t)BATCH * F2 * NP * 2);
  float*  acc   = (float*) alloc((size_t)BATCH * F2 * NP * 4);
  bf16_t* sc1   = (bf16_t*)alloc((size_t)BATCH * F1 * NP * 2);
  bf16_t* sc2   = (bf16_t*)alloc((size_t)BATCH * F2 * NP * 2);
  float*  score = (float*) alloc((size_t)BATCH * NN * 4);
  float*  mask1 = (float*) alloc((size_t)BATCH * NN * 4);
  float*  mask2 = (float*) alloc((size_t)BATCH * NN * 4);
  float*  stats = (float*) alloc((size_t)BATCH * 560 * 4);
  float*  h1    = (float*) alloc((size_t)BATCH * 500 * 4);
  float*  h2    = (float*) alloc((size_t)BATCH * 300 * 4);
  float*  h3    = (float*) alloc((size_t)BATCH * 100 * 4);

  // -------- prepare bf16 Laplacians and zero-mean input --------
  k_convert_pad<<<blocks_for((long long)NP * NP, 256), 256, 0, stream>>>(Lmat, Lbf);
  k_convert_pad<<<blocks_for((long long)NP * NP, 256), 256, 0, stream>>>(Lsmat, Lsbf);
  k_prep<<<BATCH, 256, 0, stream>>>(input, x0);

  // -------- spectral conv 1: y_k = L^k x, project onto 10 filters --------
  k_project<<<blocks_for((long long)BATCH * F1 * NN, 256), 256, 0, stream>>>(
      x0, alpha1, acc, 1, F1, 1);
  {
    const bf16_t* cur = x0;
    bf16_t* bufs[2] = {yA, yB};
    for (int k = 1; k < NTERMS; ++k) {
      bf16_t* nxt = bufs[(k - 1) & 1];
      k_gemm_wmma<<<dim3(BATCH / 512, NP / 32), 256, 0, stream>>>(Lbf, cur, nxt, nullptr, BATCH);
      k_project<<<blocks_for((long long)BATCH * F1 * NN, 256), 256, 0, stream>>>(
          nxt, alpha1 + k * 1 * F1, acc, 1, F1, 0);
      cur = nxt;
    }
  }
  k_finish<<<blocks_for((long long)BATCH * F1 * NP, 256), 256, 0, stream>>>(
      acc, beta1, nullptr, sc1, F1);

  // -------- dynamic pool 1 (top-600) --------
  k_score<<<blocks_for((long long)BATCH * NN, 256), 256, 0, stream>>>(sc1, nullptr, F1, score);
  k_topk<<<BATCH, 800, 0, stream>>>(score, nullptr, TOPK1, mask1);

  // -------- spectral conv 2 --------
  k_project<<<blocks_for((long long)BATCH * F2 * NN, 256), 256, 0, stream>>>(
      sc1, alpha2, acc, F1, F2, 1);
  {
    const bf16_t* cur = sc1;
    bf16_t* bufs[2] = {yA, yB};
    const int C = BATCH * F1;   // 5120
    for (int k = 1; k < NTERMS; ++k) {
      bf16_t* nxt = bufs[(k - 1) & 1];
      k_gemm_wmma<<<dim3(C / 512, NP / 32), 256, 0, stream>>>(Lbf, cur, nxt, nullptr, C);
      k_project<<<blocks_for((long long)BATCH * F2 * NN, 256), 256, 0, stream>>>(
          nxt, alpha2 + k * F1 * F2, acc, F1, F2, 0);
      cur = nxt;
    }
  }
  k_finish<<<blocks_for((long long)BATCH * F2 * NP, 256), 256, 0, stream>>>(
      acc, beta2, mask1, sc2, F2);

  // -------- dynamic pool 2 (top-300 within mask1) --------
  k_score<<<blocks_for((long long)BATCH * NN, 256), 256, 0, stream>>>(sc2, mask1, F2, score);
  k_topk<<<BATCH, 800, 0, stream>>>(score, mask1, TOPK2, mask2);

  // -------- Chebyshev statistic: T0..T13, moments after each term --------
  {
    const int C = BATCH * F2;   // 10240
    k_cheb_init<<<blocks_for((long long)C * NP, 256), 256, 0, stream>>>(sc2, mask2, yA);
    k_moments<<<C, 256, 0, stream>>>(yA, stats, 0);
    k_gemm_wmma<<<dim3(C / 512, NP / 32), 256, 0, stream>>>(Lsbf, yA, yB, nullptr, C);
    k_moments<<<C, 256, 0, stream>>>(yB, stats, 1);
    bf16_t* tprev = yA;   // T_{k-2}
    bf16_t* tcur  = yB;   // T_{k-1}
    for (int k = 2; k < NCHEB; ++k) {
      // T_k = 2*Ls*T_{k-1} - T_{k-2}, written in-place over T_{k-2}
      k_gemm_wmma<<<dim3(C / 512, NP / 32), 256, 0, stream>>>(Lsbf, tcur, tprev, tprev, C);
      bf16_t* newt = tprev;
      tprev = tcur;
      tcur  = newt;
      k_moments<<<C, 256, 0, stream>>>(tcur, stats, k);
    }
  }

  // -------- MLP head --------
  k_linear<<<blocks_for((long long)BATCH * 500, 256), 256, 0, stream>>>(stats, W1, b1, h1, 560, 500, 1);
  k_linear<<<blocks_for((long long)BATCH * 300, 256), 256, 0, stream>>>(h1, W2, b2, h2, 500, 300, 1);
  k_linear<<<blocks_for((long long)BATCH * 100, 256), 256, 0, stream>>>(h2, W3, b3, h3, 300, 100, 1);
  k_linear<<<blocks_for((long long)BATCH * 9,   256), 256, 0, stream>>>(h3, W4, b4, out, 100, 9, 0);
}
